// RNNModel_29394756174165
// MI455X (gfx1250) — matrix operand order repaired and checked by reading, not compiled, not run on hardware
//
#include <hip/hip_runtime.h>

typedef __bf16 bf16_t;
typedef __attribute__((ext_vector_type(16))) __bf16 v16bf;
typedef __attribute__((ext_vector_type(8)))  __bf16 v8bf;
typedef __attribute__((ext_vector_type(8)))  float  v8f;
typedef __attribute__((ext_vector_type(4)))  unsigned int v4u;
typedef __attribute__((ext_vector_type(8)))  int v8i;
typedef __attribute__((ext_vector_type(4)))  int v4i;

#define HDIM 512
#define BDIM 256
#define TDIM 512
#define DDIM 64

#if __has_builtin(__builtin_amdgcn_tensor_load_to_lds)
#define HAVE_TDM 1
#else
#define HAVE_TDM 0
#endif

static __device__ inline v16bf combine16(v8bf lo, v8bf hi) {
    v16bf r;
#pragma unroll
    for (int i = 0; i < 8; ++i) { r[i] = lo[i]; r[8 + i] = hi[i]; }
    return r;
}

// A-fragment (16x32 bf16): row = lane&15, K = {aks..aks+7, 16+aks..16+aks+7}
static __device__ inline v16bf load_a_frag_bf16(const bf16_t* rowptr, int aks) {
    const bf16_t* p = rowptr + aks;
    v8bf lo = *(const v8bf*)p;
    v8bf hi = *(const v8bf*)(p + 16);
    return combine16(lo, hi);
}

// A-fragment from fp32 source, converted on the fly
static __device__ inline v16bf load_a_frag_f32(const float* rowptr, int aks) {
    const float* p = rowptr + aks;
    v16bf r;
#pragma unroll
    for (int i = 0; i < 8; ++i) {
        r[i]     = (bf16_t)p[i];
        r[8 + i] = (bf16_t)p[16 + i];
    }
    return r;
}

// B-fragment (32x16 bf16): col = lane&15, K = {bks/2..bks/2+7, 16+bks/2..16+bks/2+7}
static __device__ inline v16bf load_b_frag(const bf16_t* W, int ldw, int nrow, int k0, int bks) {
    const bf16_t* p = W + (size_t)nrow * ldw + k0 + bks / 2;
    v8bf lo = *(const v8bf*)p;
    v8bf hi = *(const v8bf*)(p + 16);
    return combine16(lo, hi);
}

static __device__ inline v8f wmma_bf16(v16bf a, v16bf b, v8f c) {
    return __builtin_amdgcn_wmma_f32_16x16x32_bf16(
        /*neg_a=*/false, a, /*neg_b=*/false, b,
        /*c_mod=*/(short)0, c, /*reuse_a=*/false, /*reuse_b=*/false);
}

// Lane-level async DMA: 16 bytes/lane, global -> LDS (ASYNCcnt).
static __device__ inline void async_load_b128(const void* g, void* l) {
    asm volatile("global_load_async_to_lds_b128 %0, %1, off"
                 :: "v"((unsigned)(uintptr_t)l),
                    "v"((unsigned long long)(uintptr_t)g)
                 : "memory");
}
static __device__ inline void wait_async_0() {
    asm volatile("s_wait_asynccnt 0x0" ::: "memory");
}
static __device__ inline void wait_async_8() {
    asm volatile("s_wait_asynccnt 0x8" ::: "memory");
}

// TDM: one instruction copies a strided 2D tile (rows x d0 f32 elems) into LDS.
// D# built per CDNA5 ISA ch.8: group0 = {count, lds_addr, global_addr, type=2},
// group1 = {data_size=4B, tensor dims, tile dims, dim0 stride}. Tracked by
// TENSORcnt. Groups 2/3 NULL-equivalent (2D tensor).
static __device__ inline void tdm_load_tile_f32(const float* gsrc, void* ldst,
                                                unsigned d0_elems, unsigned rows,
                                                unsigned long long row_stride_elems)
{
#if HAVE_TDM
    unsigned long long ga = (unsigned long long)(uintptr_t)gsrc;
    v4u g0;
    g0.x = 1u;                                   // count=1 (valid user descriptor)
    g0.y = (unsigned)(uintptr_t)ldst;            // lds_addr
    g0.z = (unsigned)ga;                         // global_addr[31:0]
    g0.w = (unsigned)(ga >> 32) | 0x80000000u;   // global_addr[56:32] | type=2<<30
    v8i g1;
    g1[0] = (int)(2u << 16);                                       // data_size = 4B
    g1[1] = (int)(d0_elems << 16);                                 // tensor_dim0[15:0]
    g1[2] = (int)((d0_elems >> 16) | (rows << 16));                // dim0[31:16] | dim1[15:0]
    g1[3] = (int)((rows >> 16) | (d0_elems << 16));                // dim1[31:16] | tile_dim0
    g1[4] = (int)rows;                                             // tile_dim1 (tile_dim2 = 0)
    g1[5] = (int)(row_stride_elems & 0xffffffffull);               // dim0_stride[31:0]
    g1[6] = (int)((row_stride_elems >> 32) & 0xffffull);           // dim0_stride[47:32]
    g1[7] = 0;                                                     // dim1_stride unused (2D)
    v4i z4 = {0, 0, 0, 0};
#if __clang_major__ >= 23
    v8i z8 = {0, 0, 0, 0, 0, 0, 0, 0};
    __builtin_amdgcn_tensor_load_to_lds(g0, g1, z4, z4, z8, 0);
#else
    __builtin_amdgcn_tensor_load_to_lds(g0, g1, z4, z4, 0);
#endif
#endif
}

static __device__ inline void wait_tensor0() {
#if __has_builtin(__builtin_amdgcn_s_wait_tensorcnt)
    __builtin_amdgcn_s_wait_tensorcnt(0);
#else
    asm volatile("s_wait_tensorcnt 0x0" ::: "memory");
#endif
}
static __device__ inline void wait_tensor1() {
#if __has_builtin(__builtin_amdgcn_s_wait_tensorcnt)
    __builtin_amdgcn_s_wait_tensorcnt(1);
#else
    asm volatile("s_wait_tensorcnt 0x1" ::: "memory");
#endif
}

// ---------------------------------------------------------------------------
// fp32 -> bf16 conversion for weights
// ---------------------------------------------------------------------------
__global__ void cvt_f32_bf16(const float* __restrict__ src, bf16_t* __restrict__ dst, int n) {
    int i = blockIdx.x * blockDim.x + threadIdx.x;
    if (i < n) dst[i] = (bf16_t)src[i];
}

// ---------------------------------------------------------------------------
// Input-projection GEMM: C[M,512] = A[M,K] @ W^T (W row-major [512,K]) + (b0+b1)
// 256 threads = 8 waves; wave w computes rows [blk.x*128 + w*16, +16),
// cols [blk.y*64, +64). The 64xK W tile is staged into LDS once per WG via
// lane-level async-to-LDS DMA and shared by all 8 waves.
// ---------------------------------------------------------------------------
template <bool ABF16, int K>
__global__ __launch_bounds__(256) void gemm_xp_kernel(
    const void* __restrict__ Av, const bf16_t* __restrict__ W,
    const float* __restrict__ b0, const float* __restrict__ b1,
    float* __restrict__ C, int M)
{
    constexpr int LDW = K + 8;                // row pad keeps 16B alignment
    __shared__ bf16_t wt[64][LDW];

    const int wave = threadIdx.x >> 5;
    const int lane = threadIdx.x & 31;
    const int row0 = blockIdx.x * 128 + wave * 16;
    const int nb0  = blockIdx.y * 64;

    // Stage W rows [nb0, nb0+64) into LDS: K/32 async b128 ops per thread.
    {
        constexpr int UPR = K / 8;            // 16B units per W row
#pragma unroll
        for (int j = 0; j < K / 32; ++j) {
            int u   = threadIdx.x + j * 256;
            int r   = u / UPR;
            int c   = (u % UPR) * 8;
            async_load_b128(W + (size_t)(nb0 + r) * K + c, &wt[r][c]);
        }
        wait_async_0();
        __syncthreads();
    }

    const int n_lo = lane & 15;
    const int half = (lane >> 4) & 1;
    const int aks  = half * 8;
    const int bks  = half * 16;
    const int arow = row0 + n_lo;             // A row handled by this lane

    v8f zero = {0.f, 0.f, 0.f, 0.f, 0.f, 0.f, 0.f, 0.f};
    v8f acc[4] = {zero, zero, zero, zero};

    for (int k0 = 0; k0 < K; k0 += 32) {
        v16bf a;
        if (ABF16) {
            const bf16_t* A = (const bf16_t*)Av;
            a = load_a_frag_bf16(A + (size_t)arow * K + k0, aks);
        } else {
            const float* A = (const float*)Av;
            a = load_a_frag_f32(A + (size_t)arow * K + k0, aks);
        }
#pragma unroll
        for (int nt = 0; nt < 4; ++nt) {
            v16bf b = load_b_frag(&wt[0][0], LDW, nt * 16 + n_lo, k0, bks);
            acc[nt] = wmma_bf16(a, b, acc[nt]);
        }
    }

    const int mh = half * 8;
#pragma unroll
    for (int nt = 0; nt < 4; ++nt) {
        int n = nb0 + nt * 16 + n_lo;
        float bias = b0[n] + b1[n];
#pragma unroll
        for (int i = 0; i < 8; ++i) {
            C[(size_t)(row0 + mh + i) * HDIM + n] = acc[nt][i] + bias;
        }
    }
}

// ---------------------------------------------------------------------------
// Recurrence: each workgroup owns 16 batch rows, all 512 hidden columns.
// h double-buffered in LDS (bf16). xp tile for step t+1 is DMA'd into an LDS
// double buffer by the Tensor Data Mover (one tensor_load_to_lds per step,
// issued by wave 0, overlapped with the WMMA K-loop). W_hh streamed from L2.
// MODE 0: store h_t (bf16) for all t -> hout [B,T,H]
// MODE 1: store only h_{T-1} (f32)   -> hlast [B,H]
// ---------------------------------------------------------------------------
template <int MODE>
__global__ __launch_bounds__(256) void rnn_rec_kernel(
    const float* __restrict__ xp, const bf16_t* __restrict__ Whh,
    bf16_t* __restrict__ hout, float* __restrict__ hlast, int T)
{
    const int LDH = HDIM + 8;
    __shared__ bf16_t ht[2][16][HDIM + 8];    // 33 KB
    __shared__ float  xt[2][16][HDIM];        // 64 KB

    const int wave  = threadIdx.x >> 5;
    const int lane  = threadIdx.x & 31;
    const int bbase = blockIdx.x * 16;
    const int nb0   = wave * 64;

    // zero-init h_0
    for (int i = threadIdx.x; i < 16 * LDH; i += 256) {
        ht[0][i / LDH][i % LDH] = (bf16_t)0.0f;
    }
    // DMA xp tile for t = 0 (rows stride T*H floats)
#if HAVE_TDM
    if (wave == 0) {
        tdm_load_tile_f32(xp + (size_t)bbase * T * HDIM, &xt[0][0][0],
                          HDIM, 16, (unsigned long long)T * HDIM);
        wait_tensor0();
    }
#else
#pragma unroll
    for (int j = 0; j < 8; ++j) {
        int u = threadIdx.x + j * 256;
        int m = u >> 7;
        int c = (u & 127) * 4;
        async_load_b128(xp + ((size_t)(bbase + m) * T) * HDIM + c, &xt[0][m][c]);
    }
    wait_async_0();
#endif
    __syncthreads();

    const int n_lo = lane & 15;
    const int half = (lane >> 4) & 1;
    const int aks  = half * 8;
    const int bks  = half * 16;
    const int mh   = half * 8;

    v8f zero = {0.f, 0.f, 0.f, 0.f, 0.f, 0.f, 0.f, 0.f};

    for (int t = 0; t < T; ++t) {
        const int cur = t & 1, nxt = cur ^ 1;

        // kick off DMA of next step's xp tile (overlaps with WMMA below)
        if (t + 1 < T) {
#if HAVE_TDM
            if (wave == 0) {
                tdm_load_tile_f32(xp + ((size_t)bbase * T + (t + 1)) * HDIM, &xt[nxt][0][0],
                                  HDIM, 16, (unsigned long long)T * HDIM);
            }
#else
#pragma unroll
            for (int j = 0; j < 8; ++j) {
                int u = threadIdx.x + j * 256;
                int m = u >> 7;
                int c = (u & 127) * 4;
                async_load_b128(xp + ((size_t)(bbase + m) * T + (t + 1)) * HDIM + c,
                                &xt[nxt][m][c]);
            }
#endif
        }

        v8f acc[4] = {zero, zero, zero, zero};
        for (int k0 = 0; k0 < HDIM; k0 += 32) {
            v16bf a = load_a_frag_bf16(&ht[cur][n_lo][k0], aks);
            __builtin_prefetch(Whh + (size_t)(nb0 + n_lo) * HDIM + ((k0 + 64) & (HDIM - 1)), 0, 0);
#pragma unroll
            for (int nt = 0; nt < 4; ++nt) {
                v16bf b = load_b_frag(Whh, HDIM, nb0 + nt * 16 + n_lo, k0, bks);
                acc[nt] = wmma_bf16(a, b, acc[nt]);
            }
        }

        // retire this step's xp DMA (leave next step's in flight)
#if HAVE_TDM
        if (wave == 0) {
            if (t + 1 < T) wait_tensor1(); else wait_tensor0();
        }
#else
        if (t + 1 < T) wait_async_8(); else wait_async_0();
#endif
        __syncthreads();                      // xt[cur] visible to all waves

#pragma unroll
        for (int nt = 0; nt < 4; ++nt) {
            int n = nb0 + nt * 16 + n_lo;
#pragma unroll
            for (int i = 0; i < 8; ++i) {
                int m = mh + i;
                float v = tanhf(acc[nt][i] + xt[cur][m][n]);
                ht[nxt][m][n] = (bf16_t)v;
                if (MODE == 0) {
                    hout[((size_t)(bbase + m) * T + t) * HDIM + n] = (bf16_t)v;
                } else if (t == T - 1) {
                    hlast[(size_t)(bbase + m) * HDIM + n] = v;
                }
            }
        }
        __syncthreads();                      // h[nxt] written; xt[cur] free for reuse
    }
}

// ---------------------------------------------------------------------------
// Final linear: out[b] = dot(h_last[b,:], w_lin) + b_lin
// ---------------------------------------------------------------------------
__global__ __launch_bounds__(128) void final_linear_kernel(
    const float* __restrict__ hlast, const float* __restrict__ wlin,
    const float* __restrict__ blin, float* __restrict__ out)
{
    __shared__ float red[128];
    int b = blockIdx.x, tid = threadIdx.x;
    float s = 0.f;
    for (int h = tid; h < HDIM; h += 128) s += hlast[(size_t)b * HDIM + h] * wlin[h];
    red[tid] = s;
    __syncthreads();
    for (int off = 64; off > 0; off >>= 1) {
        if (tid < off) red[tid] += red[tid + off];
        __syncthreads();
    }
    if (tid == 0) out[b] = red[0] + blin[0];
}

// ---------------------------------------------------------------------------
extern "C" void kernel_launch(void* const* d_in, const int* in_sizes, int n_in,
                              void* d_out, int out_size, void* d_ws, size_t ws_size,
                              hipStream_t stream) {
    const float* x     = (const float*)d_in[0];
    const float* w_ih0 = (const float*)d_in[1];
    const float* w_hh0 = (const float*)d_in[2];
    const float* b_ih0 = (const float*)d_in[3];
    const float* b_hh0 = (const float*)d_in[4];
    const float* w_ih1 = (const float*)d_in[5];
    const float* w_hh1 = (const float*)d_in[6];
    const float* b_ih1 = (const float*)d_in[7];
    const float* b_hh1 = (const float*)d_in[8];
    const float* w_lin = (const float*)d_in[9];
    const float* b_lin = (const float*)d_in[10];

    char* ws = (char*)d_ws;
    size_t o = 0;
    bf16_t* wih0b = (bf16_t*)(ws + o); o += (size_t)HDIM * DDIM * 2;           // 64 KB
    bf16_t* whh0b = (bf16_t*)(ws + o); o += (size_t)HDIM * HDIM * 2;           // 512 KB
    bf16_t* wih1b = (bf16_t*)(ws + o); o += (size_t)HDIM * HDIM * 2;
    bf16_t* whh1b = (bf16_t*)(ws + o); o += (size_t)HDIM * HDIM * 2;
    float*  xp    = (float*)(ws + o);  o += (size_t)BDIM * TDIM * HDIM * 4;    // 256 MB
    bf16_t* h1    = (bf16_t*)(ws + o); o += (size_t)BDIM * TDIM * HDIM * 2;    // 128 MB
    float*  hlast = (float*)(ws + o);  o += (size_t)BDIM * HDIM * 4;

    // 1) weight conversion to bf16
    {
        int n0 = HDIM * DDIM, n1 = HDIM * HDIM;
        cvt_f32_bf16<<<(n0 + 255) / 256, 256, 0, stream>>>(w_ih0, wih0b, n0);
        cvt_f32_bf16<<<(n1 + 255) / 256, 256, 0, stream>>>(w_hh0, whh0b, n1);
        cvt_f32_bf16<<<(n1 + 255) / 256, 256, 0, stream>>>(w_ih1, wih1b, n1);
        cvt_f32_bf16<<<(n1 + 255) / 256, 256, 0, stream>>>(w_hh1, whh1b, n1);
    }

    const int M = BDIM * TDIM;             // 131072
    dim3 gGemm(M / 128, HDIM / 64);        // 1024 x 8

    // 2) xp0 = x @ w_ih0^T + b_ih0 + b_hh0
    gemm_xp_kernel<false, DDIM><<<gGemm, 256, 0, stream>>>(x, wih0b, b_ih0, b_hh0, xp, M);

    // 3) layer-0 recurrence -> h1 (bf16)
    rnn_rec_kernel<0><<<BDIM / 16, 256, 0, stream>>>(xp, whh0b, h1, nullptr, TDIM);

    // 4) xp1 = h1 @ w_ih1^T + b_ih1 + b_hh1 (overwrites xp buffer)
    gemm_xp_kernel<true, HDIM><<<gGemm, 256, 0, stream>>>(h1, wih1b, b_ih1, b_hh1, xp, M);

    // 5) layer-1 recurrence -> hlast (f32, last step only)
    rnn_rec_kernel<1><<<BDIM / 16, 256, 0, stream>>>(xp, whh1b, nullptr, hlast, TDIM);

    // 6) out[b] = hlast[b,:] . w_lin + b_lin
    final_linear_kernel<<<BDIM, 128, 0, stream>>>(hlast, w_lin, b_lin, (float*)d_out);
}
